// Gemma3nAttention_40346922778912
// MI455X (gfx1250) — matrix operand order repaired and checked by reading, not compiled, not run on hardware
//
#include <hip/hip_runtime.h>
#include <hip/hip_bf16.h>
#include <cstdint>
#include <cstddef>

typedef __bf16 bf16;
typedef bf16 v16bf __attribute__((ext_vector_type(16)));
typedef bf16 v8bf  __attribute__((ext_vector_type(8)));
typedef bf16 v4bf  __attribute__((ext_vector_type(4)));
typedef float v8f  __attribute__((ext_vector_type(8)));
typedef unsigned int v4u __attribute__((ext_vector_type(4)));
typedef int v8i __attribute__((ext_vector_type(8)));
typedef int v4i __attribute__((ext_vector_type(4)));

#define B_   2
#define S_   2048
#define HID_ 2048
#define H_   8
#define KV_  2
#define D_   256
#define SWIN 512

#if defined(__has_builtin)
#if __has_builtin(__builtin_amdgcn_tensor_load_to_lds) && __has_builtin(__builtin_amdgcn_s_wait_tensorcnt)
#define HAVE_TDM 1
#endif
#endif
#ifndef HAVE_TDM
#define HAVE_TDM 0
#endif

// ---------------- WMMA fragment helpers (CDNA5 16-bit layouts) ----------------
// A-matrix 16x32: lane L holds row m=L%16; K = {kb8..kb8+7} U {kb8+16..kb8+23}, kb8=(L/16)*8
__device__ __forceinline__ v16bf frag_a(const bf16* p) {
  v8bf lo = *(const v8bf*)p;
  v8bf hi = *(const v8bf*)(p + 16);
  v16bf r;
#pragma unroll
  for (int i = 0; i < 8; ++i) { r[i] = lo[i]; r[i + 8] = hi[i]; }
  return r;
}
// B-matrix 32x16: lane L holds col n=L%16; K = {kb8..kb8+7} U {kb8+16..kb8+23}, kb8=(L/16)*8
__device__ __forceinline__ v16bf frag_b(const bf16* p) {
  v8bf lo = *(const v8bf*)p;
  v8bf hi = *(const v8bf*)(p + 16);
  v16bf r;
#pragma unroll
  for (int i = 0; i < 8; ++i) { r[i] = lo[i]; r[i + 8] = hi[i]; }
  return r;
}
#define WMMA_BF16(a, b, c) \
  __builtin_amdgcn_wmma_f32_16x16x32_bf16(false, (a), false, (b), (short)0, (c), false, false)

// ---------------- fp32 -> bf16 bulk convert ----------------
__global__ __launch_bounds__(256) void cvt_f32_bf16(const float* __restrict__ x,
                                                    bf16* __restrict__ y, int n) {
  int i = (blockIdx.x * 256 + threadIdx.x) * 4;
  if (i + 3 < n) {
    const float4 v = *(const float4*)(x + i);
    v4bf o;
    o[0] = (bf16)v.x; o[1] = (bf16)v.y; o[2] = (bf16)v.z; o[3] = (bf16)v.w;
    *(v4bf*)(y + i) = o;
  }
}

// ---------------- GEMM: Y[M,N] = A[M,K](bf16) * W[N,K](f32)^T ----------------
// Block tile 64x128, 8 waves as 2x4, wave tile 32x32 (2x2 WMMA), K step 32.
// A tile (64x32 bf16) is DMA'd by the Tensor Data Mover into padded LDS rows
// (TDM pad: 4 DWORDs after every 16 DWORDs -> 40-element rows, bank-conflict free).
__global__ __launch_bounds__(256) void gemm_bf16_xwT(const bf16* __restrict__ A,
                                                     const float* __restrict__ W,
                                                     void* __restrict__ Yout,
                                                     int M, int N, int K, int out_f32) {
  __shared__ bf16 As[64][40];   // 5120 B == one TDM tile incl. padding
  __shared__ bf16 Bs[128][40];
  const int m0 = blockIdx.y * 64;
  const int n0 = blockIdx.x * 128;
  const int t = threadIdx.x;
  const int wave = t >> 5, lane = t & 31;
  const int wm = wave >> 2, wn = wave & 3;
  const int lm = lane & 15, lh = lane >> 4;
  v8f acc[2][2] = {};

  const int br = t >> 1, bc = (t & 1) * 16;  // W: 128 rows x 32 cols, 16 f32/thread
#if !HAVE_TDM
  const int ar = t >> 2, ac = (t & 3) * 8;   // fallback manual A path
#endif

  for (int k0 = 0; k0 < K; k0 += 32) {
#if !HAVE_TDM
    v8bf va = *(const v8bf*)(A + (size_t)(m0 + ar) * K + k0 + ac);
#endif
    const float* wsrc = W + (size_t)(n0 + br) * K + k0 + bc;
    v8bf wb0, wb1;
#pragma unroll
    for (int i = 0; i < 8; ++i) {
      wb0[i] = (bf16)wsrc[i];
      wb1[i] = (bf16)wsrc[8 + i];
    }
    __syncthreads();  // previous iteration's LDS reads done before overwrite

#if HAVE_TDM
    if (wave == 0) {
      // Tensor DMA descriptor (CDNA5 ISA): 2D bf16 tile
      const uint64_t ga =
          (uint64_t)(uintptr_t)A + ((uint64_t)m0 * (uint64_t)K + (uint64_t)k0) * 2ull;
      const uint32_t ldsa = (uint32_t)(uintptr_t)(&As[0][0]);  // low 32b = LDS offset
      v4u g0;
      g0[0] = 1u;                                   // count=1 (valid user D#)
      g0[1] = ldsa;                                 // lds_addr
      g0[2] = (uint32_t)(ga & 0xFFFFFFFFu);         // global_addr[31:0]
      g0[3] = (uint32_t)((ga >> 32) & 0x01FFFFFFu)  // global_addr[56:32]
              | (2u << 30);                         // type = 2 ("image")
      v8i g1;
      g1[0] = (int)((1u << 16)      // data_size = 2 bytes
                    | (1u << 20)    // pad_enable
                    | (3u << 22)    // pad_interval: 16 DWORDs (64B row)
                    | (3u << 25));  // pad_amount:   4 DWORDs (16B pad)
      g1[1] = (int)(((uint32_t)K & 0xFFFFu) << 16);                     // tensor_dim0 lo
      g1[2] = (int)((((uint32_t)K >> 16) & 0xFFFFu) |
                    (((uint32_t)M & 0xFFFFu) << 16));                   // dim0 hi | dim1 lo
      g1[3] = (int)((((uint32_t)M >> 16) & 0xFFFFu) | (32u << 16));     // dim1 hi | tile_dim0=32
      g1[4] = (int)64u;                                                 // tile_dim1=64, tile_dim2=0
      g1[5] = (int)(uint32_t)K;                                         // tensor_dim0_stride lo
      g1[6] = 0;
      g1[7] = 0;
      v4i gz = {0, 0, 0, 0};
#if defined(__clang_major__) && (__clang_major__ >= 23)
      v8i gz8 = {0, 0, 0, 0, 0, 0, 0, 0};
      __builtin_amdgcn_tensor_load_to_lds(g0, g1, gz, gz, gz8, 0);
#else
      __builtin_amdgcn_tensor_load_to_lds(g0, g1, gz, gz, 0);
#endif
    }
#else
    *(v8bf*)(&As[ar][ac]) = va;
#endif
    *(v8bf*)(&Bs[br][bc]) = wb0;
    *(v8bf*)(&Bs[br][bc + 8]) = wb1;
#if HAVE_TDM
    if (wave == 0) __builtin_amdgcn_s_wait_tensorcnt(0);  // TENSORcnt == 0
#endif
    __syncthreads();
#pragma unroll
    for (int mi = 0; mi < 2; ++mi) {
      v16bf af = frag_a(&As[wm * 32 + mi * 16 + lm][lh * 8]);
#pragma unroll
      for (int ni = 0; ni < 2; ++ni) {
        v16bf bfr = frag_b(&Bs[wn * 32 + ni * 16 + lm][lh * 8]);
        acc[mi][ni] = WMMA_BF16(af, bfr, acc[mi][ni]);
      }
    }
  }
  // C/D layout: vgpr r -> row r + 8*(lane/16), col = lane%16
#pragma unroll
  for (int mi = 0; mi < 2; ++mi)
#pragma unroll
    for (int ni = 0; ni < 2; ++ni)
#pragma unroll
      for (int r = 0; r < 8; ++r) {
        int row = m0 + wm * 32 + mi * 16 + lh * 8 + r;
        int col = n0 + wn * 32 + ni * 16 + lm;
        float v = acc[mi][ni][r];
        if (out_f32) ((float*)Yout)[(size_t)row * N + col] = v;
        else ((bf16*)Yout)[(size_t)row * N + col] = (bf16)v;
      }
}

// ---------------- RoPE + unit LayerNorm (one wave per 256-elem head row) ----------------
// Lane data map: d = (lane/16)*128 + (lane%16)*8 + i, i=0..7  -> rope partner via shfl_xor 16.
__global__ __launch_bounds__(256) void rope_norm_kernel(const bf16* __restrict__ Yq,
                                                        const bf16* __restrict__ Yk,
                                                        const bf16* __restrict__ Yv,
                                                        bf16* __restrict__ Qb,
                                                        bf16* __restrict__ Kb,
                                                        bf16* __restrict__ Vt) {
  const int wid = blockIdx.x * 8 + (threadIdx.x >> 5);
  const int lane = threadIdx.x & 31;
  const int lm = lane & 15, lh = lane >> 4;
  const int dbase = lh * 128 + lm * 8;

  const bf16* src;
  bf16* dst;
  size_t dst_base;
  bool do_rope, strided;
  float outscale;
  int s;

  if (wid < B_ * S_ * H_) {
    int h = wid % H_; int tt = wid / H_; s = tt % S_; int b = tt / S_;
    src = Yq + (((size_t)(b * S_ + s)) * H_ + h) * D_;
    do_rope = true; strided = false; outscale = 256.0f;  // query_pre_attn_scalar
    dst = Qb;
    dst_base = (((size_t)(b * H_ + h)) * S_ + s) * D_ + dbase;
  } else if (wid < B_ * S_ * (H_ + KV_)) {
    int w = wid - B_ * S_ * H_;
    int h = w % KV_; int tt = w / KV_; s = tt % S_; int b = tt / S_;
    src = Yk + (((size_t)(b * S_ + s)) * KV_ + h) * D_;
    do_rope = true; strided = false; outscale = 1.0f;
    dst = Kb;
    dst_base = (((size_t)(b * KV_ + h)) * S_ + s) * D_ + dbase;
  } else {
    int w = wid - B_ * S_ * (H_ + KV_);
    int h = w % KV_; int tt = w / KV_; s = tt % S_; int b = tt / S_;
    src = Yv + (((size_t)(b * S_ + s)) * KV_ + h) * D_;
    do_rope = false; strided = true; outscale = 1.0f;   // V: norm only, store transposed [D][S]
    dst = Vt;
    dst_base = (((size_t)(b * KV_ + h)) * D_ + dbase) * (size_t)S_ + s;
  }

  v8bf xr = *(const v8bf*)(src + dbase);
  float y[8];
#pragma unroll
  for (int i = 0; i < 8; ++i) y[i] = (float)xr[i];

  if (do_rope) {
    const float pos = (float)s;
#pragma unroll
    for (int i = 0; i < 8; ++i) {
      float part = __shfl_xor(y[i], 16);             // x[d+-128]
      float rot = (lh == 0) ? -part : part;          // rotate_half
      int dmod = lm * 8 + i;                         // d % 128
      float invf = __powf(1000000.0f, -(float)dmod * (1.0f / 128.0f));
      float ang = pos * invf;
      float sn, cs;
      __sincosf(ang, &sn, &cs);
      y[i] = y[i] * cs + rot * sn;
    }
  }
  // unit LayerNorm over 256 (full-wave reduction)
  float sum = 0.0f;
#pragma unroll
  for (int i = 0; i < 8; ++i) sum += y[i];
#pragma unroll
  for (int m = 1; m < 32; m <<= 1) sum += __shfl_xor(sum, m);
  float mu = sum * (1.0f / 256.0f);
  float var = 0.0f;
#pragma unroll
  for (int i = 0; i < 8; ++i) { float d = y[i] - mu; var += d * d; }
#pragma unroll
  for (int m = 1; m < 32; m <<= 1) var += __shfl_xor(var, m);
  float rs = rsqrtf(var * (1.0f / 256.0f) + 1e-6f);

  if (!strided) {
    v8bf o;
#pragma unroll
    for (int i = 0; i < 8; ++i) o[i] = (bf16)((y[i] - mu) * rs * outscale);
    *(v8bf*)(dst + dst_base) = o;
  } else {
#pragma unroll
    for (int i = 0; i < 8; ++i)
      dst[dst_base + (size_t)i * S_] = (bf16)((y[i] - mu) * rs);
  }
}

// ---------------- Sliding-window flash attention ----------------
// grid = (S/128, B*H); 8 waves per block, each owns a 16-query tile.
__global__ __launch_bounds__(256) void attn_kernel(const bf16* __restrict__ Qb,
                                                   const bf16* __restrict__ Kb,
                                                   const bf16* __restrict__ Vt,
                                                   bf16* __restrict__ Ob) {
  __shared__ bf16 P[8][16][32];  // per-wave probability tile (C-layout -> A-layout bridge)
  const int qblk = blockIdx.x;
  const int bh = blockIdx.y;
  const int b = bh >> 3, h = bh & 7, kvh = h >> 2;  // GQA: 4 q-heads per kv-head
  const int wave = threadIdx.x >> 5, lane = threadIdx.x & 31;
  const int lm = lane & 15, lh = lane >> 4;
  const int q0b = qblk * 128;
  const int q0 = q0b + wave * 16;

  const bf16* Q = Qb + (((size_t)(b * H_ + h)) * S_ + q0) * D_;
  const bf16* K = Kb + ((size_t)(b * KV_ + kvh)) * S_ * D_;
  const bf16* V = Vt + ((size_t)(b * KV_ + kvh)) * D_ * S_;  // [D][S]

  v8f O[16] = {};
  float mstate[8], lstate[8];
#pragma unroll
  for (int r = 0; r < 8; ++r) { mstate[r] = -3.0e38f; lstate[r] = 0.0f; }

  int lo = q0b - SWIN + 1;
  if (lo < 0) lo = 0;
  lo &= ~31;
  const int hiblk = q0b + 96;  // last 32-key block start covering q0b+127

  for (int kb = lo; kb <= hiblk; kb += 32) {
    // ---- scores: S[16q x 32k] = Q[16 x 256] * K^T, 16 WMMAs ----
    v8f s1 = {}, s2 = {};
#pragma unroll
    for (int dk = 0; dk < 8; ++dk) {
      v16bf qa = frag_a(Q + (size_t)lm * D_ + dk * 32 + lh * 8);
      v16bf k1 = frag_b(K + (size_t)(kb + lm) * D_ + dk * 32 + lh * 8);
      v16bf k2 = frag_b(K + (size_t)(kb + 16 + lm) * D_ + dk * 32 + lh * 8);
      s1 = WMMA_BF16(qa, k1, s1);
      s2 = WMMA_BF16(qa, k2, s2);
    }
    if (kb + 32 <= hiblk)
      __builtin_prefetch(K + (size_t)(kb + 32 + lm) * D_, 0, 1);  // global_prefetch_b8

    // ---- softcap + mask + online softmax (per-row, rows split by lane-half) ----
    float alpha[8];
#pragma unroll
    for (int r = 0; r < 8; ++r) {
      const int qi = q0 + lh * 8 + r;
      float a = s1[r] * 0.0625f;                 // / sqrt(256)
      a = 30.0f * tanhf(a * (1.0f / 30.0f));     // softcap
      int j1 = kb + lm;
      a = ((j1 <= qi) && (j1 > qi - SWIN)) ? a : -30.0f;  // tanh(-inf)*30
      float c = s2[r] * 0.0625f;
      c = 30.0f * tanhf(c * (1.0f / 30.0f));
      int j2 = kb + 16 + lm;
      c = ((j2 <= qi) && (j2 > qi - SWIN)) ? c : -30.0f;
      float rm = fmaxf(a, c);
      rm = fmaxf(rm, __shfl_xor(rm, 1));
      rm = fmaxf(rm, __shfl_xor(rm, 2));
      rm = fmaxf(rm, __shfl_xor(rm, 4));
      rm = fmaxf(rm, __shfl_xor(rm, 8));
      float mnew = fmaxf(mstate[r], rm);
      float al = __expf(mstate[r] - mnew);
      float p1 = __expf(a - mnew);
      float p2 = __expf(c - mnew);
      float rsum = p1 + p2;
      rsum += __shfl_xor(rsum, 1);
      rsum += __shfl_xor(rsum, 2);
      rsum += __shfl_xor(rsum, 4);
      rsum += __shfl_xor(rsum, 8);
      lstate[r] = lstate[r] * al + rsum;
      mstate[r] = mnew;
      alpha[r] = al;
      P[wave][lh * 8 + r][lm] = (bf16)p1;
      P[wave][lh * 8 + r][16 + lm] = (bf16)p2;
    }
#pragma unroll
    for (int t = 0; t < 16; ++t)
#pragma unroll
      for (int r = 0; r < 8; ++r) O[t][r] *= alpha[r];

    __syncthreads();  // P visible (uniform trip count across waves)

    // ---- O[16 x 256] += P[16 x 32] * V[32 x 256], 16 WMMAs; V is [D][S] ----
    v16bf pa = frag_a(&P[wave][lm][lh * 8]);
#pragma unroll
    for (int t = 0; t < 16; ++t) {
      v16bf vb = frag_b(V + (size_t)(t * 16 + lm) * S_ + kb + lh * 8);
      O[t] = WMMA_BF16(pa, vb, O[t]);
    }
    __syncthreads();
  }

  // ---- normalize + store [B][S][H*D] ----
#pragma unroll
  for (int t = 0; t < 16; ++t)
#pragma unroll
    for (int r = 0; r < 8; ++r) {
      int row = q0 + lh * 8 + r;
      int d = t * 16 + lm;
      float v = O[t][r] / lstate[r];
      Ob[((size_t)b * S_ + row) * HID_ + h * D_ + d] = (bf16)v;
    }
}

// ---------------- host orchestration ----------------
extern "C" void kernel_launch(void* const* d_in, const int* in_sizes, int n_in,
                              void* d_out, int out_size, void* d_ws, size_t ws_size,
                              hipStream_t stream) {
  const float* hidden = (const float*)d_in[0];
  const float* Wq = (const float*)d_in[1];
  const float* Wk = (const float*)d_in[2];
  const float* Wv = (const float*)d_in[3];
  const float* Wo = (const float*)d_in[4];

  char* ws = (char*)d_ws;
  size_t off = 0;
  auto alloc = [&](size_t bytes) {
    void* p = ws + off;
    off += (bytes + 255) & ~(size_t)255;
    return p;
  };
  const size_t M = (size_t)B_ * S_;  // 4096
  bf16* Xbf = (bf16*)alloc(M * HID_ * 2);          // hidden in bf16
  bf16* Yq = (bf16*)alloc(M * (H_ * D_) * 2);      // q proj [B,S,H,D]
  bf16* Yk = (bf16*)alloc(M * (KV_ * D_) * 2);     // k proj [B,S,KV,D]
  bf16* Yv = (bf16*)alloc(M * (KV_ * D_) * 2);     // v proj [B,S,KV,D]
  bf16* Qb = (bf16*)alloc((size_t)B_ * H_ * S_ * D_ * 2);   // [B,H,S,D]
  bf16* Kb = (bf16*)alloc((size_t)B_ * KV_ * S_ * D_ * 2);  // [B,KV,S,D]
  bf16* Vt = (bf16*)alloc((size_t)B_ * KV_ * D_ * S_ * 2);  // [B,KV,D,S]
  bf16* Ab = (bf16*)alloc(M * HID_ * 2);           // attn out [B,S,H*D]

  cvt_f32_bf16<<<8192, 256, 0, stream>>>(hidden, Xbf, (int)(M * HID_));

  gemm_bf16_xwT<<<dim3(2048 / 128, 4096 / 64), 256, 0, stream>>>(
      Xbf, Wq, Yq, 4096, 2048, 2048, 0);
  gemm_bf16_xwT<<<dim3(512 / 128, 4096 / 64), 256, 0, stream>>>(
      Xbf, Wk, Yk, 4096, 512, 2048, 0);
  gemm_bf16_xwT<<<dim3(512 / 128, 4096 / 64), 256, 0, stream>>>(
      Xbf, Wv, Yv, 4096, 512, 2048, 0);

  // B*S*(H+2*KV) = 49152 head-rows, 8 waves/block -> 6144 blocks
  rope_norm_kernel<<<6144, 256, 0, stream>>>(Yq, Yk, Yv, Qb, Kb, Vt);

  attn_kernel<<<dim3(S_ / 128, B_ * H_), 256, 0, stream>>>(Qb, Kb, Vt, Ab);

  gemm_bf16_xwT<<<dim3(2048 / 128, 4096 / 64), 256, 0, stream>>>(
      Ab, Wo, d_out, 4096, 2048, 2048, 1);
}
